// Base_9191230013898
// MI455X (gfx1250) — matrix operand order repaired and checked by reading, not compiled, not run on hardware
//
#include <hip/hip_runtime.h>
#include <stdint.h>

// ---------------------------------------------------------------------------
// GNN message passing, restructured for MI455X (gfx1250):
//   per-edge GEMM  ->  per-node GEMM (bf16x3-split WMMA) + bandwidth-only edge op
// FLOPs: ~1.6e11 (GEMMs), edge stream ~2.5 GB total -> ~107us memory floor at
// 23.3 TB/s; bf16 WMMA keeps GEMM time at/below that floor while the bf16
// hi/lo 3-product split preserves ~fp32 accuracy vs the fp32 reference.
// ---------------------------------------------------------------------------

#define N_NODES 100000
#define N_EDGES 200000
#define N_LAYERS 3
#define DIM 300      // node feature dim
#define HID 150      // per-direction message dim
#define KPAD 320     // K (=300) padded to multiple of 32 for 16x16x32 WMMA

typedef __attribute__((ext_vector_type(16))) __bf16 v16bf;
typedef __attribute__((ext_vector_type(8)))  __bf16 v8bf;
typedef __attribute__((ext_vector_type(8)))  float  v8f;

#define WMMA_BF16(A, B, C) \
  __builtin_amdgcn_wmma_f32_16x16x32_bf16(false, (A), false, (B), (short)0, (C), false, false)

__device__ __forceinline__ float gelu_exact(float z) {
  return 0.5f * z * (1.0f + erff(z * 0.70710678118654752440f));
}

// RNE fp32 -> bf16 (bits), plus residual low half for the bf16x3 split.
__device__ __forceinline__ void split_bf16_pair(float f, uint16_t& hi, uint16_t& lo) {
  uint32_t u = __float_as_uint(f);
  uint32_t hb = (u + 0x7FFFu + ((u >> 16) & 1u)) >> 16;
  float fh = __uint_as_float(hb << 16);
  float fl = f - fh;
  uint32_t ul = __float_as_uint(fl);
  uint32_t lb = (ul + 0x7FFFu + ((ul >> 16) & 1u)) >> 16;
  hi = (uint16_t)hb;
  lo = (uint16_t)lb;
}

// --- fragment loaders (layouts per CDNA5 ISA §7.12.2) -------------
// A 16x32 bf16: lane (m = lane&15, g = lane>>4) holds k = g*8 + {0..7} and
//               16 + g*8 + {0..7}  -> two contiguous 16B chunks.
__device__ __forceinline__ v16bf load_a_frag(const uint16_t* rowp, int k0, int g) {
  union { v16bf v; v8bf h[2]; } u;
  u.h[0] = *(const v8bf*)(rowp + k0 + g * 8);
  u.h[1] = *(const v8bf*)(rowp + k0 + 16 + g * 8);
  return u.v;
}
// B 32x16 bf16 (weights stored transposed [Nout][Kpad]): lane (n = lane&15,
// g = lane>>4) holds k = g*8 + {0..7} and 16 + g*8 + {0..7} -> two 16B chunks.
__device__ __forceinline__ v16bf load_b_frag(const uint16_t* colp, int k0, int g) {
  union { v16bf v; v8bf h[2]; } u;
  u.h[0] = *(const v8bf*)(colp + k0 + g * 8);
  u.h[1] = *(const v8bf*)(colp + k0 + 16 + g * 8);
  return u.v;
}

// ---------------------------------------------------------------------------
// Split fp32 [M,Kin] into hi/lo bf16 [M,KPAD] (zero K-pad).
__global__ void k_split(const float* __restrict__ in, uint16_t* __restrict__ hi,
                        uint16_t* __restrict__ lo, int M, int Kin) {
  int tid = blockIdx.x * blockDim.x + threadIdx.x;
  int m = tid / KPAD, k = tid - m * KPAD;
  if (m >= M) return;
  float f = (k < Kin) ? in[(size_t)m * Kin + k] : 0.0f;
  uint16_t h, l;
  split_bf16_pair(f, h, l);
  hi[(size_t)m * KPAD + k] = h;
  lo[(size_t)m * KPAD + k] = l;
}

// Build Wcat^T [600][KPAD] = [Wr_top | Wr_bot | Wc_top | Wc_bot]^T as bf16 hi/lo.
// Wr/Wc layer slice is row-major [600][150] (rows = 2D input dim, cols = H).
__global__ void k_prep_wcat(const float* __restrict__ Wr, const float* __restrict__ Wc,
                            uint16_t* __restrict__ hi, uint16_t* __restrict__ lo) {
  int tid = blockIdx.x * blockDim.x + threadIdx.x;
  int n = tid / KPAD, k = tid - n * KPAD;
  if (n >= 2 * DIM) return;
  float f = 0.0f;
  if (k < DIM) {
    if (n < HID)            f = Wr[(size_t)k * HID + n];                    // Wr_top
    else if (n < 2 * HID)   f = Wr[(size_t)(DIM + k) * HID + (n - HID)];    // Wr_bot
    else if (n < 3 * HID)   f = Wc[(size_t)k * HID + (n - 2 * HID)];        // Wc_top
    else                    f = Wc[(size_t)(DIM + k) * HID + (n - 3 * HID)];// Wc_bot
  }
  uint16_t h, l;
  split_bf16_pair(f, h, l);
  hi[(size_t)n * KPAD + k] = h;
  lo[(size_t)n * KPAD + k] = l;
}

// Build Wa^T [300][KPAD] bf16 hi/lo. Wa row-major [300][300].
__global__ void k_prep_wa(const float* __restrict__ Wa, uint16_t* __restrict__ hi,
                          uint16_t* __restrict__ lo) {
  int tid = blockIdx.x * blockDim.x + threadIdx.x;
  int n = tid / KPAD, k = tid - n * KPAD;
  if (n >= DIM) return;
  float f = (k < DIM) ? Wa[(size_t)k * DIM + n] : 0.0f;
  uint16_t h, l;
  split_bf16_pair(f, h, l);
  hi[(size_t)n * KPAD + k] = h;
  lo[(size_t)n * KPAD + k] = l;
}

// ---------------------------------------------------------------------------
// bf16x3-split GEMM: C[M,Nout] = A[M,300] @ B[300,Nout], A given as hi/lo bf16
// [M,KPAD], B^T given as hi/lo bf16 [Nout,KPAD].
// bias==nullptr : plain store.  bias!=nullptr : C = resid + GELU(acc + bias).
// Block 256 = 8 waves, each wave a 32x32 tile; block covers 128(M) x 64(N).
__global__ __launch_bounds__(256) void k_gemm_bf16x3(
    const uint16_t* __restrict__ Ahi, const uint16_t* __restrict__ Alo,
    const uint16_t* __restrict__ Bhi, const uint16_t* __restrict__ Blo,
    float* __restrict__ Cout, const float* __restrict__ bias,
    const float* __restrict__ resid, int M, int Nout) {
  const int lane = threadIdx.x & 31;
  const int wave = threadIdx.x >> 5;
  const int g = lane >> 4;
  const int ln = lane & 15;
  const int baseM = blockIdx.x * 128 + (wave & 3) * 32;
  const int baseN = blockIdx.y * 64 + (wave >> 2) * 32;

  int rA[2], cB[2];
  rA[0] = baseM + ln;       if (rA[0] > M - 1) rA[0] = M - 1;
  rA[1] = baseM + 16 + ln;  if (rA[1] > M - 1) rA[1] = M - 1;
  cB[0] = baseN + ln;       if (cB[0] > Nout - 1) cB[0] = Nout - 1;
  cB[1] = baseN + 16 + ln;  if (cB[1] > Nout - 1) cB[1] = Nout - 1;

  const uint16_t* aHiP[2] = { Ahi + (size_t)rA[0] * KPAD, Ahi + (size_t)rA[1] * KPAD };
  const uint16_t* aLoP[2] = { Alo + (size_t)rA[0] * KPAD, Alo + (size_t)rA[1] * KPAD };
  const uint16_t* bHiP[2] = { Bhi + (size_t)cB[0] * KPAD, Bhi + (size_t)cB[1] * KPAD };
  const uint16_t* bLoP[2] = { Blo + (size_t)cB[0] * KPAD, Blo + (size_t)cB[1] * KPAD };

  v8f acc[2][2] = {};

#pragma unroll
  for (int k0 = 0; k0 < KPAD; k0 += 32) {
    v16bf ah[2], al[2], bh[2], bl[2];
#pragma unroll
    for (int i = 0; i < 2; ++i) {
      ah[i] = load_a_frag(aHiP[i], k0, g);
      al[i] = load_a_frag(aLoP[i], k0, g);
      bh[i] = load_b_frag(bHiP[i], k0, g);
      bl[i] = load_b_frag(bLoP[i], k0, g);
    }
#pragma unroll
    for (int i = 0; i < 2; ++i)
#pragma unroll
      for (int j = 0; j < 2; ++j) {
        acc[i][j] = WMMA_BF16(ah[i], bh[j], acc[i][j]);  // hi*hi
        acc[i][j] = WMMA_BF16(al[i], bh[j], acc[i][j]);  // lo*hi
        acc[i][j] = WMMA_BF16(ah[i], bl[j], acc[i][j]);  // hi*lo
      }
  }

#pragma unroll
  for (int i = 0; i < 2; ++i)
#pragma unroll
    for (int j = 0; j < 2; ++j) {
      const int col = baseN + j * 16 + ln;
      if (col >= Nout) continue;
      const float b = bias ? bias[col] : 0.0f;
#pragma unroll
      for (int r = 0; r < 8; ++r) {
        const int row = baseM + i * 16 + r + 8 * g;
        if (row >= M) continue;
        float v = acc[i][j][r];
        if (bias) v = resid[(size_t)row * Nout + col] + gelu_exact(v + b);
        Cout[(size_t)row * Nout + col] = v;
      }
    }
}

// ---------------------------------------------------------------------------
// Edge stage: proj[n] = [v@Wr_top | v@Wr_bot | v@Wc_top | v@Wc_bot]  ([N,600]).
//   r-msg at edge (s->d): GELU(proj[d][h] + proj[s][150+h] + br[h]) -> RC[d][h]
//   c-msg (flipped)     : GELU(proj[s][300+h] + proj[d][450+h] + bc[h]) -> RC[s][150+h]
// One thread per (edge, h); consecutive h -> coalesced loads/atomics.
__global__ void k_edges(const int* __restrict__ ei, const float* __restrict__ proj,
                        const float* __restrict__ br, const float* __restrict__ bc,
                        float* __restrict__ RC) {
  int tid = blockIdx.x * blockDim.x + threadIdx.x;
  int e = tid / HID, h = tid - e * HID;
  if (e >= N_EDGES) return;
  int s = ei[e];
  int d = ei[N_EDGES + e];
  const float* ps = proj + (size_t)s * (4 * HID);
  const float* pd = proj + (size_t)d * (4 * HID);
  float zr = pd[h] + ps[HID + h] + br[h];
  atomicAdd(&RC[(size_t)d * DIM + h], gelu_exact(zr));
  float zc = ps[2 * HID + h] + pd[3 * HID + h] + bc[h];
  atomicAdd(&RC[(size_t)s * DIM + HID + h], gelu_exact(zc));
}

// ---------------------------------------------------------------------------
extern "C" void kernel_launch(void* const* d_in, const int* in_sizes, int n_in,
                              void* d_out, int out_size, void* d_ws, size_t ws_size,
                              hipStream_t stream) {
  const float* x  = (const float*)d_in[0];
  const int*   ei = (const int*)d_in[1];
  const float* Wr = (const float*)d_in[2];
  const float* br = (const float*)d_in[3];
  const float* Wc = (const float*)d_in[4];
  const float* bc = (const float*)d_in[5];
  const float* Wa = (const float*)d_in[6];
  const float* ba = (const float*)d_in[7];
  float* out = (float*)d_out;

  // Workspace carve-up (all region sizes are multiples of 256 B). ~730 MB.
  char* p = (char*)d_ws;
  float*    vA     = (float*)p;    p += (size_t)N_NODES * DIM * 4;        // 120 MB
  float*    vB     = (float*)p;    p += (size_t)N_NODES * DIM * 4;        // 120 MB
  float*    proj   = (float*)p;    p += (size_t)N_NODES * 4 * HID * 4;    // 240 MB
  float*    RC     = (float*)p;    p += (size_t)N_NODES * DIM * 4;        // 120 MB
  uint16_t* Ahi    = (uint16_t*)p; p += (size_t)N_NODES * KPAD * 2;       // 64 MB
  uint16_t* Alo    = (uint16_t*)p; p += (size_t)N_NODES * KPAD * 2;       // 64 MB
  uint16_t* WcatHi = (uint16_t*)p; p += (size_t)2 * DIM * KPAD * 2;
  uint16_t* WcatLo = (uint16_t*)p; p += (size_t)2 * DIM * KPAD * 2;
  uint16_t* WaHi   = (uint16_t*)p; p += (size_t)DIM * KPAD * 2;
  uint16_t* WaLo   = (uint16_t*)p; p += (size_t)DIM * KPAD * 2;

  const int TB = 256;
  const int splitBlocks = (N_NODES * KPAD + TB - 1) / TB;
  const int edgeBlocks  = (N_EDGES * HID + TB - 1) / TB;
  const dim3 gProj((N_NODES + 127) / 128, (2 * DIM + 63) / 64);  // N x 600
  const dim3 gUpd((N_NODES + 127) / 128, (DIM + 63) / 64);       // N x 300

  // Wa^T is layer-invariant: prep once per call.
  k_prep_wa<<<(DIM * KPAD + TB - 1) / TB, TB, 0, stream>>>(Wa, WaHi, WaLo);

  const float* vin = x;
  for (int l = 0; l < N_LAYERS; ++l) {
    // 1) split node features into bf16 hi/lo
    k_split<<<splitBlocks, TB, 0, stream>>>(vin, Ahi, Alo, N_NODES, DIM);
    // 2) per-layer concatenated projection weights, transposed
    k_prep_wcat<<<(2 * DIM * KPAD + TB - 1) / TB, TB, 0, stream>>>(
        Wr + (size_t)l * 2 * DIM * HID, Wc + (size_t)l * 2 * DIM * HID, WcatHi, WcatLo);
    // 3) proj = v @ Wcat  (N x 300 x 600, WMMA)
    k_gemm_bf16x3<<<gProj, TB, 0, stream>>>(Ahi, Alo, WcatHi, WcatLo, proj,
                                            nullptr, nullptr, N_NODES, 2 * DIM);
    // 4) zero aggregation buffer, then edge messages + scatter-add
    hipMemsetAsync(RC, 0, (size_t)N_NODES * DIM * 4, stream);
    k_edges<<<edgeBlocks, TB, 0, stream>>>(ei, proj, br + (size_t)l * HID,
                                           bc + (size_t)l * HID, RC);
    // 5) node update: v_out = v_in + GELU(RC @ Wa + ba)  (N x 300 x 300, WMMA)
    k_split<<<splitBlocks, TB, 0, stream>>>(RC, Ahi, Alo, N_NODES, DIM);
    float* vout = (l == N_LAYERS - 1) ? out : ((l == 0) ? vA : vB);
    k_gemm_bf16x3<<<gUpd, TB, 0, stream>>>(Ahi, Alo, WaHi, WaLo, vout,
                                           ba, vin, N_NODES, DIM);
    vin = vout;
  }
}
